// SelfAttention_65481071397372
// MI455X (gfx1250) — matrix operand order repaired and checked by reading, not compiled, not run on hardware
//
#include <hip/hip_runtime.h>
#include <stdint.h>

// ---------------------------------------------------------------------------
// Problem constants (reference: E=512, H=128, B=8, T=2048)
// ---------------------------------------------------------------------------
#define E_DIM 512
#define H_DIM 128
#define B_SZ  8
#define T_SZ  2048
#define BT    (B_SZ * T_SZ)   // 16384 total rows

typedef __attribute__((ext_vector_type(16))) __bf16       v16bf;
typedef __attribute__((ext_vector_type(8)))  float        v8f;
typedef __attribute__((ext_vector_type(4)))  unsigned int v4u;

// float -> bf16 (round-to-nearest-even) bit pattern
__device__ __forceinline__ unsigned short f2bf(float f) {
    unsigned int u = __float_as_uint(f);
    u = (u + 0x7FFFu + ((u >> 16) & 1u)) >> 16;
    return (unsigned short)u;
}

union FragU {
    v16bf v;
    v4u   q[2];
};

// A fragment (16x32 bf16, row m = lane%16).
// lane<16 : K = {kk+0..7, kk+16..23};  lane>=16 : K = {kk+8..15, kk+24..31}
__device__ __forceinline__ v16bf load_frag_a(const unsigned short* rowp, int kk, int sel) {
    FragU f;
    f.q[0] = *(const v4u*)(rowp + kk + sel * 8);
    f.q[1] = *(const v4u*)(rowp + kk + 16 + sel * 8);
    return f.v;
}

// B fragment (32x16 bf16, column n = lane%16)
// lane<16 : K = {kk+0..7, kk+16..23};  lane>=16 : K = {kk+8..15, kk+24..31}
__device__ __forceinline__ v16bf load_frag_b(const unsigned short* rowp, int kk, int sel) {
    FragU f;
    f.q[0] = *(const v4u*)(rowp + kk + sel * 8);
    f.q[1] = *(const v4u*)(rowp + kk + 16 + sel * 8);
    return f.v;
}

__device__ __forceinline__ v8f wmma_bf16(v16bf a, v16bf b, v8f c) {
    // (neg_a, A, neg_b, B, c_mod, C, reuse_a, reuse_b)
    return __builtin_amdgcn_wmma_f32_16x16x32_bf16(false, a, false, b, (short)0, c, false, false);
}

#define V8F_ZERO ((v8f){0.f,0.f,0.f,0.f,0.f,0.f,0.f,0.f})

// ---------------------------------------------------------------------------
// Kernel 1: x (f32) -> xb (bf16)
// ---------------------------------------------------------------------------
__global__ void cvt_x_kernel(const float* __restrict__ x, unsigned short* __restrict__ xb, int n) {
    int i = blockIdx.x * blockDim.x + threadIdx.x;
    int stride = gridDim.x * blockDim.x;
    for (; i < n; i += stride) xb[i] = f2bf(x[i]);
}

// ---------------------------------------------------------------------------
// Kernel 2: Wq/Wk/Wv (E,E) and W1 (E,H) -> transposed bf16 (row = output col)
// ---------------------------------------------------------------------------
__global__ void cvt_w_kernel(const float* __restrict__ Wq, const float* __restrict__ Wk,
                             const float* __restrict__ Wv, const float* __restrict__ W1,
                             unsigned short* __restrict__ WqT, unsigned short* __restrict__ WkT,
                             unsigned short* __restrict__ WvT, unsigned short* __restrict__ W1T) {
    const int NW = E_DIM * E_DIM;           // 262144
    const int TOT = 3 * NW + H_DIM * E_DIM; // + 65536
    int i = blockIdx.x * blockDim.x + threadIdx.x;
    int stride = gridDim.x * blockDim.x;
    for (; i < TOT; i += stride) {
        if (i < 3 * NW) {
            int wi = i / NW, r = i % NW;
            int f = r / E_DIM, e = r % E_DIM;          // r = f*E + e
            const float* W = (wi == 0) ? Wq : (wi == 1) ? Wk : Wv;
            unsigned short* WT = (wi == 0) ? WqT : (wi == 1) ? WkT : WvT;
            WT[r] = f2bf(W[e * E_DIM + f]);
        } else {
            int r = i - 3 * NW;
            int h = r / E_DIM, e = r % E_DIM;          // r = h*E + e
            W1T[r] = f2bf(W1[e * H_DIM + h]);
        }
    }
}

// ---------------------------------------------------------------------------
// Kernel 3: q = x@Wq, k = x@Wk, v = x@Wv  (bf16 in, f32 accum, bf16 out)
// One wave per 32(t) x 16(f) tile of all three outputs (6 WMMAs / K-step,
// 10 b128 loads / K-step), software-pipelined (double-buffered) operands.
// q,k stored row-major; v stored transposed (vT[b][f][t]).
// ---------------------------------------------------------------------------
__global__ void __launch_bounds__(128)
qkv_kernel(const unsigned short* __restrict__ xb,
           const unsigned short* __restrict__ WqT,
           const unsigned short* __restrict__ WkT,
           const unsigned short* __restrict__ WvT,
           unsigned short* __restrict__ q,
           unsigned short* __restrict__ k,
           unsigned short* __restrict__ vT) {
    const int lane = threadIdx.x & 31;
    const int wave = threadIdx.x >> 5;
    const int gw = blockIdx.x * 4 + wave;      // 0 .. (BT/32)*(E/16)-1 = 16383
    const int tpair = gw >> 5;                 // / (E_DIM/16): 0..511
    const int ftile = gw & 31;
    const int t0 = tpair * 32;
    const int f0 = ftile * 16;
    const int sel = lane >> 4;
    const int ln  = lane & 15;

    const unsigned short* arow0 = xb  + (size_t)(t0 + ln) * E_DIM;
    const unsigned short* arow1 = xb  + (size_t)(t0 + 16 + ln) * E_DIM;
    const unsigned short* bqrow = WqT + (size_t)(f0 + ln) * E_DIM;
    const unsigned short* bkrow = WkT + (size_t)(f0 + ln) * E_DIM;
    const unsigned short* bvrow = WvT + (size_t)(f0 + ln) * E_DIM;

    v8f aq0 = V8F_ZERO, ak0 = V8F_ZERO, av0 = V8F_ZERO;
    v8f aq1 = V8F_ZERO, ak1 = V8F_ZERO, av1 = V8F_ZERO;

    // software pipeline: preload K-step 0
    v16bf a0 = load_frag_a(arow0, 0, sel);
    v16bf a1 = load_frag_a(arow1, 0, sel);
    v16bf bq = load_frag_b(bqrow, 0, sel);
    v16bf bk = load_frag_b(bkrow, 0, sel);
    v16bf bv = load_frag_b(bvrow, 0, sel);

    #pragma unroll 4
    for (int kk = 0; kk < E_DIM; kk += 32) {
        const int kn = (kk + 32) & (E_DIM - 1);     // wraps to 0 on last iter
        v16bf a0n = load_frag_a(arow0, kn, sel);
        v16bf a1n = load_frag_a(arow1, kn, sel);
        v16bf bqn = load_frag_b(bqrow, kn, sel);
        v16bf bkn = load_frag_b(bkrow, kn, sel);
        v16bf bvn = load_frag_b(bvrow, kn, sel);

        aq0 = wmma_bf16(a0, bq, aq0);
        aq1 = wmma_bf16(a1, bq, aq1);
        ak0 = wmma_bf16(a0, bk, ak0);
        ak1 = wmma_bf16(a1, bk, ak1);
        av0 = wmma_bf16(a0, bv, av0);
        av1 = wmma_bf16(a1, bv, av1);

        a0 = a0n; a1 = a1n; bq = bqn; bk = bkn; bv = bvn;
    }

    #pragma unroll
    for (int r = 0; r < 8; ++r) {
        const int M  = r + sel * 8;            // D layout: lane>=16 -> M = r+8
        const int f  = f0 + ln;
        // first row-tile
        int tg = t0 + M;
        q[(size_t)tg * E_DIM + f] = f2bf(aq0[r]);
        k[(size_t)tg * E_DIM + f] = f2bf(ak0[r]);
        vT[((size_t)(tg >> 11) * E_DIM + f) * T_SZ + (tg & (T_SZ - 1))] = f2bf(av0[r]);
        // second row-tile
        tg = t0 + 16 + M;
        q[(size_t)tg * E_DIM + f] = f2bf(aq1[r]);
        k[(size_t)tg * E_DIM + f] = f2bf(ak1[r]);
        vT[((size_t)(tg >> 11) * E_DIM + f) * T_SZ + (tg & (T_SZ - 1))] = f2bf(av1[r]);
    }
}

// ---------------------------------------------------------------------------
// Kernel 4: fused attention (scores -> softmax -> @V) + MLP head.
// One block (4 waves / 128 threads) per (batch, 16-query-row) tile.
// LDS: scores f32 128KB | P bf16 64KB | out bf16 16KB | hidden f32 8KB = 216KB
// ---------------------------------------------------------------------------
#define SMEM_S 0
#define SMEM_P (16 * T_SZ * 4)                     // 131072
#define SMEM_O (SMEM_P + 16 * T_SZ * 2)            // 196608
#define SMEM_H (SMEM_O + 16 * E_DIM * 2)           // 212992
#define SMEM_TOTAL (SMEM_H + 16 * H_DIM * 4)       // 221184

__global__ void __launch_bounds__(128)
attn_kernel(const unsigned short* __restrict__ q,
            const unsigned short* __restrict__ k,
            const unsigned short* __restrict__ vT,
            const unsigned short* __restrict__ W1T,
            const float* __restrict__ b1,
            const float* __restrict__ W2,
            const float* __restrict__ b2,
            float* __restrict__ out) {
    extern __shared__ char smem[];
    float*          sS = (float*)(smem + SMEM_S);          // [16][T_SZ]
    unsigned short* sP = (unsigned short*)(smem + SMEM_P); // [16][T_SZ]
    unsigned short* sO = (unsigned short*)(smem + SMEM_O); // [16][E_DIM]
    float*          sH = (float*)(smem + SMEM_H);          // [16][H_DIM]

    const int b    = blockIdx.x >> 7;          // / (T_SZ/16)
    const int t0g  = blockIdx.x * 16;          // global query row
    const int lane = threadIdx.x & 31;
    const int wave = threadIdx.x >> 5;
    const int sel  = lane >> 4;
    const int ln   = lane & 15;
    const float scale = 0.04419417382415922f;  // 1/sqrt(512)

    // ---- Phase 1: scores = (q kT)/sqrt(E) into LDS (f32) -------------------
    // The whole 16x512 q tile A-operand lives in registers (16 frags = 128
    // VGPRs) and is reused for all 128 s-tiles; inner loop is B-load + WMMA
    // only, with the B stream double-buffered.
    {
        const unsigned short* qrow = q + (size_t)(t0g + ln) * E_DIM;
        v16bf afrag[16];
        #pragma unroll
        for (int t = 0; t < 16; ++t) afrag[t] = load_frag_a(qrow, t * 32, sel);

        for (int stile = wave; stile < T_SZ / 16; stile += 4) {
            const int s0 = stile * 16;
            const unsigned short* krow = k + (size_t)(b * T_SZ + s0 + ln) * E_DIM;
            // prefetch this wave's next s-tile k row (global_prefetch_b8)
            if (stile + 4 < T_SZ / 16)
                __builtin_prefetch(k + (size_t)(b * T_SZ + s0 + 64 + ln) * E_DIM, 0, 0);

            v8f acc = V8F_ZERO;
            v16bf bcur = load_frag_b(krow, 0, sel);
            #pragma unroll
            for (int t = 0; t < 16; ++t) {
                v16bf bnxt = load_frag_b(krow, ((t + 1) * 32) & (E_DIM - 1), sel);
                acc = wmma_bf16(afrag[t], bcur, acc);
                bcur = bnxt;
            }
            #pragma unroll
            for (int r = 0; r < 8; ++r) {
                const int M = r + sel * 8;
                sS[M * T_SZ + s0 + ln] = acc[r] * scale;
            }
        }
    }
    __syncthreads();

    // ---- Phase 2: row softmax in LDS, write P as bf16 ----------------------
    {
        const int row = threadIdx.x >> 3;      // 8 threads per row
        const int j   = threadIdx.x & 7;
        float* srow = sS + row * T_SZ;
        float m = -3.4e38f;
        for (int c = j; c < T_SZ; c += 8) m = fmaxf(m, srow[c]);
        for (int off = 1; off < 8; off <<= 1) m = fmaxf(m, __shfl_xor(m, off, 8));
        float sum = 0.f;
        for (int c = j; c < T_SZ; c += 8) {
            float e = __expf(srow[c] - m);
            srow[c] = e;
            sum += e;
        }
        for (int off = 1; off < 8; off <<= 1) sum += __shfl_xor(sum, off, 8);
        const float inv = 1.0f / sum;
        unsigned short* prow = sP + row * T_SZ;
        for (int c = j; c < T_SZ; c += 8) prow[c] = f2bf(srow[c] * inv);
    }
    __syncthreads();

    // ---- Phase 3: out = P @ V ; wave w owns f-columns [w*128, w*128+128) ---
    {
        const unsigned short* arow = sP + ln * T_SZ;
        const unsigned short* brow[8];
        #pragma unroll
        for (int nt = 0; nt < 8; ++nt) {
            const int f = wave * 128 + nt * 16 + ln;
            brow[nt] = vT + ((size_t)b * E_DIM + f) * T_SZ;
        }
        v8f acc[8];
        #pragma unroll
        for (int nt = 0; nt < 8; ++nt) acc[nt] = V8F_ZERO;

        v16bf a = load_frag_a(arow, 0, sel);   // A from LDS, double-buffered
        for (int ss = 0; ss < T_SZ; ss += 32) {
            v16bf an = load_frag_a(arow, (ss + 32) & (T_SZ - 1), sel);
            // issue the 8 independent B loads first, then the 8 WMMAs, so the
            // scheduler can retire them with partial load waits
            v16bf bf[8];
            #pragma unroll
            for (int nt = 0; nt < 8; ++nt) bf[nt] = load_frag_b(brow[nt], ss, sel);
            #pragma unroll
            for (int nt = 0; nt < 8; ++nt) acc[nt] = wmma_bf16(a, bf[nt], acc[nt]);
            a = an;
        }
        #pragma unroll
        for (int nt = 0; nt < 8; ++nt) {
            #pragma unroll
            for (int r = 0; r < 8; ++r) {
                const int M = r + sel * 8;
                const int f = wave * 128 + nt * 16 + ln;
                sO[M * E_DIM + f] = f2bf(acc[nt][r]);
            }
        }
    }
    __syncthreads();

    // ---- Phase 4: h = relu(out @ W1 + b1); wave w owns h-tiles 2w, 2w+1 ----
    {
        const unsigned short* arow = sO + ln * E_DIM;
        const unsigned short* brow0 = W1T + (size_t)(wave * 32 + ln) * E_DIM;
        const unsigned short* brow1 = W1T + (size_t)(wave * 32 + 16 + ln) * E_DIM;
        v8f acc0 = V8F_ZERO, acc1 = V8F_ZERO;
        v16bf a  = load_frag_a(arow, 0, sel);
        v16bf b0 = load_frag_b(brow0, 0, sel);
        v16bf b1f = load_frag_b(brow1, 0, sel);
        #pragma unroll 4
        for (int kk = 0; kk < E_DIM; kk += 32) {
            const int kn = (kk + 32) & (E_DIM - 1);
            v16bf an  = load_frag_a(arow, kn, sel);
            v16bf b0n = load_frag_b(brow0, kn, sel);
            v16bf b1n = load_frag_b(brow1, kn, sel);
            acc0 = wmma_bf16(a, b0, acc0);
            acc1 = wmma_bf16(a, b1f, acc1);
            a = an; b0 = b0n; b1f = b1n;
        }
        #pragma unroll
        for (int r = 0; r < 8; ++r) {
            const int M = r + sel * 8;
            int h = wave * 32 + ln;
            float v0 = acc0[r] + b1[h];
            sH[M * H_DIM + h] = v0 > 0.f ? v0 : 0.f;
            h += 16;
            float v1 = acc1[r] + b1[h];
            sH[M * H_DIM + h] = v1 > 0.f ? v1 : 0.f;
        }
    }
    __syncthreads();

    // ---- Phase 5: logits = h @ W2 + b2 (16x2, scalar) ----------------------
    if (threadIdx.x < 32) {
        const int m = threadIdx.x >> 1;
        const int o = threadIdx.x & 1;
        float sum = b2[o];
        #pragma unroll 8
        for (int hh = 0; hh < H_DIM; ++hh)
            sum += sH[m * H_DIM + hh] * W2[hh * 2 + o];
        out[(size_t)(t0g + m) * 2 + o] = sum;
    }
}

// ---------------------------------------------------------------------------
// Host launcher. Workspace layout (bytes), needs ~68.9 MB:
//   xb 16MB | q 16MB | k 16MB | vT 16MB | WqT/WkT/WvT 512KB ea | W1T 128KB
// ---------------------------------------------------------------------------
extern "C" void kernel_launch(void* const* d_in, const int* in_sizes, int n_in,
                              void* d_out, int out_size, void* d_ws, size_t ws_size,
                              hipStream_t stream) {
    (void)in_sizes; (void)n_in; (void)out_size; (void)ws_size;
    const float* x  = (const float*)d_in[0];
    const float* Wq = (const float*)d_in[1];
    const float* Wk = (const float*)d_in[2];
    const float* Wv = (const float*)d_in[3];
    const float* W1 = (const float*)d_in[4];
    const float* b1 = (const float*)d_in[5];
    const float* W2 = (const float*)d_in[6];
    const float* b2 = (const float*)d_in[7];
    float* out = (float*)d_out;

    char* ws = (char*)d_ws;
    const size_t SZ_MAT = (size_t)BT * E_DIM * 2;     // 16 MB (bf16)
    unsigned short* xb  = (unsigned short*)(ws);
    unsigned short* qb  = (unsigned short*)(ws + SZ_MAT);
    unsigned short* kb  = (unsigned short*)(ws + 2 * SZ_MAT);
    unsigned short* vTb = (unsigned short*)(ws + 3 * SZ_MAT);
    unsigned short* WqT = (unsigned short*)(ws + 4 * SZ_MAT);
    unsigned short* WkT = (unsigned short*)(ws + 4 * SZ_MAT + 1 * 524288);
    unsigned short* WvT = (unsigned short*)(ws + 4 * SZ_MAT + 2 * 524288);
    unsigned short* W1T = (unsigned short*)(ws + 4 * SZ_MAT + 3 * 524288);

    cvt_x_kernel<<<4096, 256, 0, stream>>>(x, xb, BT * E_DIM);
    cvt_w_kernel<<<1024, 256, 0, stream>>>(Wq, Wk, Wv, W1, WqT, WkT, WvT, W1T);

    // (BT/32) * (E/16) tiles, 4 waves per block
    qkv_kernel<<<(BT / 32) * (E_DIM / 16) / 4, 128, 0, stream>>>(
        xb, WqT, WkT, WvT, qb, kb, vTb);

    // One block per (batch, 16-row) query tile; 216 KB dynamic LDS (<320 KB/WGP)
    (void)hipFuncSetAttribute(reinterpret_cast<const void*>(attn_kernel),
                              hipFuncAttributeMaxDynamicSharedMemorySize,
                              (int)SMEM_TOTAL);
    attn_kernel<<<BT / 16, 128, SMEM_TOTAL, stream>>>(
        qb, kb, vTb, W1T, b1, W2, b2, out);
}
